// HiddenPf_21938692948135
// MI455X (gfx1250) — matrix operand order repaired and checked by reading, not compiled, not run on hardware
//
#include <hip/hip_runtime.h>
#include <math.h>

// Problem constants (match reference)
#define B_    64
#define NSPIN 64     // N
#define TWO_N 128
#define NH_   64
#define NT_   128    // Pfaffian matrix size
#define HID   2048   // MLP hidden width
#define F1N   8192   // NH * TWO_N
#define F2N   2016   // NH*(NH-1)/2
#define SLABW 128    // N columns staged per block in kernel 2

typedef __attribute__((ext_vector_type(16))) __bf16  v16bf;
typedef __attribute__((ext_vector_type(8)))  float   v8f;

// A-matrix (16x32, MxK): lanes 0-15 hold K = e, lanes 16-31 hold K = 16 + e
// (contiguous order, matching B).
__device__ __forceinline__ int a_kmap(int e, int lane) {
  return e + ((lane >> 4) << 4);
}
// B-matrix (32x16, KxN): lanes 0-15 hold K = e, lanes 16-31 hold K = 16 + e.

// Async global->LDS 16-byte copy (CDNA5, tracked by ASYNCcnt).
__device__ __forceinline__ void async_ld_b128(unsigned lds_off, const float* g) {
  asm volatile("global_load_async_to_lds_b128 %0, %1, off"
               :: "v"(lds_off), "v"((unsigned long long)(uintptr_t)g)
               : "memory");
}
__device__ __forceinline__ void wait_asynccnt0() {
  asm volatile("s_wait_asynccnt 0x0" ::: "memory");
}

// -------------------------------------------------------------------------
// Kernel 1: H{1,2}[64,2048] = tanh(X[64,64] @ W{1,2}a[64,2048] + b) -> bf16
// -------------------------------------------------------------------------
__global__ __launch_bounds__(256) void hidden_kernel(
    const float* __restrict__ x,
    const float* __restrict__ W1a, const float* __restrict__ b1a,
    const float* __restrict__ W2a, const float* __restrict__ b2a,
    __bf16* __restrict__ H1, __bf16* __restrict__ H2) {
  int wid  = (blockIdx.x * blockDim.x + threadIdx.x) >> 5;
  int lane = threadIdx.x & 31;
  const int NTILES = HID / 16;  // 128
  if (wid >= 2 * NTILES) return;
  int sel = (wid >= NTILES) ? 1 : 0;
  int n0  = (sel ? wid - NTILES : wid) << 4;
  const float* W    = sel ? W2a : W1a;
  const float* bias = sel ? b2a : b1a;
  __bf16* Hout      = sel ? H2  : H1;
  int col = lane & 15, hi = lane >> 4;

  v8f acc[4] = {};
  for (int kt = 0; kt < NSPIN / 32; ++kt) {
    int k0 = kt << 5;
    v16bf bfrag;
#pragma unroll
    for (int e = 0; e < 16; ++e)
      bfrag[e] = (__bf16)W[(k0 + e + (hi << 4)) * HID + n0 + col];
#pragma unroll
    for (int mt = 0; mt < 4; ++mt) {
      v16bf afrag;
      int row = (mt << 4) + col;
#pragma unroll
      for (int e = 0; e < 16; ++e)
        afrag[e] = (__bf16)x[row * NSPIN + k0 + a_kmap(e, lane)];
      acc[mt] = __builtin_amdgcn_wmma_f32_16x16x32_bf16(
          false, afrag, false, bfrag, (short)0, acc[mt], false, false);
    }
  }
  float bv = bias[n0 + col];
#pragma unroll
  for (int mt = 0; mt < 4; ++mt)
#pragma unroll
    for (int v = 0; v < 8; ++v) {
      int m = (mt << 4) + v + (hi << 3);
      Hout[m * HID + n0 + col] = (__bf16)tanhf(acc[mt][v] + bv);
    }
}

// -------------------------------------------------------------------------
// Kernel 2: F1[64,8192] = H1 @ W1b + b1b ; F2[64,2016] = H2 @ W2b + b2b
// Block = 8 waves owning a 128-col weight slab; 32x128 f32 W tile staged in
// LDS via async global->LDS b128 (double-buffered, 1 barrier / K-step).
// LDS 16B chunks XOR-swizzled so the two wave-halves of each ds_load_b32 hit
// disjoint bank groups. Blocks 0..63 -> F1, 64..79 -> F2 (tail guarded).
// -------------------------------------------------------------------------
__global__ __launch_bounds__(256) void biggemm_kernel(
    const __bf16* __restrict__ H1, const __bf16* __restrict__ H2,
    const float* __restrict__ W1b, const float* __restrict__ b1b,
    const float* __restrict__ W2b, const float* __restrict__ b2b,
    float* __restrict__ F1, float* __restrict__ F2) {
  __shared__ float Wbuf[2][32][SLABW];   // 32 KB, two 16 KB tiles

  int tid  = threadIdx.x;
  int lane = tid & 31;
  int wv   = tid >> 5;                   // wave id in block = n-tile in slab
  int blk  = blockIdx.x;
  bool s2  = blk >= (F1N / SLABW);       // 64 F1 blocks, 16 F2 blocks
  const __bf16* A    = s2 ? H2  : H1;
  const float*  W    = s2 ? W2b : W1b;
  const float*  bias = s2 ? b2b : b1b;
  float*        outp = s2 ? F2  : F1;
  int ncols = s2 ? F2N : F1N;
  int nblk  = (s2 ? blk - F1N / SLABW : blk) * SLABW;
  int col = lane & 15, hi = lane >> 4;
  int n0  = nblk + (wv << 4);
  bool active = (n0 + 16) <= ncols;      // tiles are 16-wide; F2N % 16 == 0

  // issue 4 async 16B chunks per thread for K-step kt into Wbuf[bufsel]
  auto issue = [&](int kt, int bufsel) {
    int k0 = kt << 5;
#pragma unroll
    for (int i = 0; i < 4; ++i) {
      int c    = tid + (i << 8);         // chunk id 0..1023
      int row  = c >> 5;                 // 0..31 (K within tile)
      int cw   = c & 31;                 // chunk within row
      int gcol = nblk + (cw << 2);
      if (gcol + 4 > ncols) gcol = ncols - 4;          // F2 tail clamp
      int sc   = cw ^ ((row >> 1) & 15);               // bank swizzle
      unsigned lds = (unsigned)(uintptr_t)&Wbuf[bufsel][row][sc << 2];
      const float* g = W + (size_t)(k0 + row) * ncols + gcol;
      async_ld_b128(lds, g);
    }
  };

  v8f acc[4] = {};
  issue(0, 0);
  for (int kt = 0; kt < HID / 32; ++kt) {  // 64 K-steps
    wait_asynccnt0();
    __syncthreads();                       // Wbuf[kt&1] ready block-wide;
                                           // prior reads of other buffer done
    if (kt + 1 < HID / 32) issue(kt + 1, (kt + 1) & 1);

    int k0 = kt << 5;
    v16bf bfrag;
#pragma unroll
    for (int e = 0; e < 16; ++e) {
      int row  = e + (hi << 4);
      int wcol = (wv << 4) + col;
      int sc   = (wcol >> 2) ^ ((row >> 1) & 15);
      bfrag[e] = (__bf16)Wbuf[kt & 1][row][(sc << 2) + (wcol & 3)];
    }
#pragma unroll
    for (int mt = 0; mt < 4; ++mt) {
      v16bf afrag;
      int rowm = (mt << 4) + col;
#pragma unroll
      for (int e = 0; e < 16; ++e)
        afrag[e] = A[rowm * HID + k0 + a_kmap(e, lane)];
      acc[mt] = __builtin_amdgcn_wmma_f32_16x16x32_bf16(
          false, afrag, false, bfrag, (short)0, acc[mt], false, false);
    }
  }

  float bv = active ? bias[n0 + col] : 0.f;
  if (active) {
#pragma unroll
    for (int mt = 0; mt < 4; ++mt)
#pragma unroll
      for (int v = 0; v < 8; ++v) {
        int m = (mt << 4) + v + (hi << 3);
        outp[m * ncols + n0 + col] = acc[mt][v] + bv;
      }
  }
}

// -------------------------------------------------------------------------
// Kernel 3: per-batch Pfaffian (Parlett-Reid, partial pivoting) in LDS.
// -------------------------------------------------------------------------
__device__ __forceinline__ float Mval(int i, int j, const int* fidx,
    const float* F_vec, const float* F1b, const float* F2b) {
  if (i < NSPIN) {
    if (j < NSPIN) {
      int fi = fidx[i], fj = fidx[j];
      return (fi > fj) ? F_vec[(fi * (fi - 1)) / 2 + fj] : 0.f;
    }
    return 0.f;
  }
  if (j < NSPIN) return F1b[(i - NSPIN) * TWO_N + fidx[j]];
  int h = i - NSPIN, g = j - NSPIN;
  return (h > g) ? F2b[(h * (h - 1)) / 2 + g] : 0.f;
}

__global__ __launch_bounds__(512) void pfaffian_kernel(
    const float* __restrict__ x, const float* __restrict__ F_vec,
    const float* __restrict__ F1, const float* __restrict__ F2,
    float* __restrict__ out) {
  __shared__ float As[NT_][NT_ + 1];  // stride 129 -> conflict-free columns
  __shared__ float vs[NT_];
  __shared__ float rv[NT_];
  __shared__ int   ri[NT_];
  __shared__ int   fidx[NSPIN];

  int tid = threadIdx.x;
  int b   = blockIdx.x;
  const float* xi  = x  + b * NSPIN;
  const float* F1b = F1 + b * F1N;
  const float* F2b = F2 + b * F2N;

  if (tid == 0) {
    int cnt = 0;
    for (int t = 0; t < TWO_N; ++t) {
      bool occ = (t < NSPIN) ? (xi[t] > 0.f) : (xi[t - NSPIN] <= 0.f);
      if (occ) fidx[cnt++] = t;
    }
  }
  __syncthreads();

  const float scale = sqrtf(2.71828182845904523536f / (float)NT_);
  for (int idx = tid; idx < NT_ * NT_; idx += 512) {
    int i = idx >> 7, j = idx & 127;
    float mij = Mval(i, j, fidx, F_vec, F1b, F2b);
    float mji = Mval(j, i, fidx, F_vec, F1b, F2b);
    As[i][j] = scale * (mij - mji);
  }
  __syncthreads();

  float sign = 1.f;
  for (int jc = 0; jc < NT_ - 2; ++jc) {
    // argmax_i>jc |A[i][jc]| (tree reduction; on ties not always the first index)
    if (tid < NT_) {
      rv[tid] = (tid > jc) ? fabsf(As[tid][jc]) : -1.f;
      ri[tid] = tid;
    }
    __syncthreads();
    for (int s = NT_ / 2; s > 0; s >>= 1) {
      if (tid < s) {
        if (rv[tid + s] > rv[tid]) { rv[tid] = rv[tid + s]; ri[tid] = ri[tid + s]; }
      }
      __syncthreads();
    }
    int p = ri[0];
    int q = jc + 1;
    if (p != q) {                                // uniform branch
      sign = -sign;
      if (tid < NT_) { float t0 = As[p][tid]; As[p][tid] = As[q][tid]; As[q][tid] = t0; }
      __syncthreads();
      if (tid < NT_) { float t0 = As[tid][p]; As[tid][p] = As[tid][q]; As[tid][q] = t0; }
      __syncthreads();
    }
    float piv = As[q][jc];
    if (tid < NT_) vs[tid] = (tid > q) ? As[tid][jc] / piv : 0.f;
    __syncthreads();
    // Fused rank-2 update; row q / col q provably invariant, so skip them.
    for (int idx = tid; idx < NT_ * NT_; idx += 512) {
      int r = idx >> 7, c = idx & 127;
      if (r != q && c != q) {
        float t1 = vs[r] * As[q][c];
        float t2 = As[q][r] * vs[c];
        As[r][c] = As[r][c] - (t1 - t2);
      }
    }
    __syncthreads();
  }

  if (tid == 0) {
    float prod = sign;
    for (int i = 0; i < NT_; i += 2) prod *= As[i][i + 1];
    out[b] = prod;
  }
}

// -------------------------------------------------------------------------
extern "C" void kernel_launch(void* const* d_in, const int* in_sizes, int n_in,
                              void* d_out, int out_size, void* d_ws, size_t ws_size,
                              hipStream_t stream) {
  (void)in_sizes; (void)n_in; (void)out_size; (void)ws_size;
  const float* x   = (const float*)d_in[0];
  const float* Fv  = (const float*)d_in[1];
  const float* W1a = (const float*)d_in[2];
  const float* b1a = (const float*)d_in[3];
  const float* W1b = (const float*)d_in[4];
  const float* b1b = (const float*)d_in[5];
  const float* W2a = (const float*)d_in[6];
  const float* b2a = (const float*)d_in[7];
  const float* W2b = (const float*)d_in[8];
  const float* b2b = (const float*)d_in[9];
  float* out = (float*)d_out;

  // Workspace layout: H1(bf16) | H2(bf16) | F1(f32) | F2(f32)  (~3.1 MB)
  char* ws = (char*)d_ws;
  __bf16* H1 = (__bf16*)ws;
  __bf16* H2 = (__bf16*)(ws + (size_t)B_ * HID * 2);
  float*  F1 = (float*)(ws + (size_t)2 * B_ * HID * 2);
  float*  F2 = (float*)(ws + (size_t)2 * B_ * HID * 2 + (size_t)B_ * F1N * 4);

  hidden_kernel<<<32, 256, 0, stream>>>(x, W1a, b1a, W2a, b2a, H1, H2);

  int blocks2 = F1N / SLABW + (F2N + SLABW - 1) / SLABW;  // 64 + 16 = 80
  biggemm_kernel<<<blocks2, 256, 0, stream>>>(H1, H2, W1b, b1b, W2b, b2b, F1, F2);

  pfaffian_kernel<<<B_, 512, 0, stream>>>(x, Fv, F1, F2, out);
}
